// CondCliffordSteerableKernel_53644141527682
// MI455X (gfx1250) — matrix operand order repaired and checked by reading, not compiled, not run on hardware
//
#include <hip/hip_runtime.h>
#include <cstdint>
#include <cstddef>

typedef _Float16 h16;
typedef __attribute__((ext_vector_type(16))) _Float16 v16h;
typedef __attribute__((ext_vector_type(8)))  float    v8f;

// ---------------- compile-time Clifford tables (DIM=3, Euclidean) -------------
// Blade order mirrors the reference: (), (0),(1),(2), (01),(02),(12), (012)
struct CayT {
  int   prod[8][8];   // blade index of basis product a*b
  float sign[8][8];   // sign of basis product a*b
  int   grade[8];
  int   pid[8][8];    // path id for (grade(a), grade(a*b), grade(b))
  int   mask[8];      // blade index -> bitmask
  int   m2i[8];       // bitmask -> blade index
};

constexpr CayT buildCay() {
  CayT c{};
  const int masks[8] = {0, 1, 2, 4, 3, 5, 6, 7};
  for (int i = 0; i < 8; ++i) c.mask[i] = masks[i];
  for (int i = 0; i < 8; ++i) c.m2i[masks[i]] = i;
  for (int i = 0; i < 8; ++i) {
    int g = 0;
    for (int b = 0; b < 3; ++b) g += (masks[i] >> b) & 1;
    c.grade[i] = g;
  }
  for (int a = 0; a < 8; ++a)
    for (int b = 0; b < 8; ++b) {
      const int ma = masks[a], mb = masks[b];
      int sw = 0;
      for (int j = 0; j < 3; ++j)
        if ((mb >> j) & 1)
          for (int t = j + 1; t < 3; ++t) sw += (ma >> t) & 1;
      c.prod[a][b] = c.m2i[ma ^ mb];
      c.sign[a][b] = (sw & 1) ? -1.0f : 1.0f;
    }
  bool gpf[4][4][4] = {};
  for (int a = 0; a < 8; ++a)
    for (int b = 0; b < 8; ++b)
      gpf[c.grade[a]][c.grade[c.prod[a][b]]][c.grade[b]] = true;
  int pidt[4][4][4] = {};
  int p = 0;
  for (int x = 0; x < 4; ++x)
    for (int y = 0; y < 4; ++y)
      for (int z = 0; z < 4; ++z) pidt[x][y][z] = gpf[x][y][z] ? p++ : 20;
  for (int a = 0; a < 8; ++a)
    for (int b = 0; b < 8; ++b)
      c.pid[a][b] = pidt[c.grade[a]][c.grade[c.prod[a][b]]][c.grade[b]];
  return c;
}
constexpr CayT CAY = buildCay();

// ---------------- problem constants ----------------
constexpr int   NG     = 343;                     // 7^3 grid points
constexpr int   NPAD   = 352;                     // padded n stride for kf_t
constexpr int   HH     = 64;                      // hidden channels
constexpr float FACTOR = 0.05400617248673217f;    // 1/sqrt(343)

// workspace layout (bytes)
constexpr size_t OFF_LIN0F = 0;                       //  32 frags * 1KB
constexpr size_t OFF_LINSF = OFF_LIN0F + 32  * 1024;  //  96 frags
constexpr size_t OFF_GPSF  = OFF_LINSF + 96  * 1024;  // 640 frags
constexpr size_t OFF_OUTF  = OFF_GPSF  + 640 * 1024;  // 512 frags
constexpr size_t OFF_KFT   = OFF_OUTF  + 512 * 1024;  // 8192*352 floats
constexpr size_t WS_NEED   = OFF_KFT + (size_t)8192 * NPAD * 4;

// ---------------- WMMA helpers ----------------
__device__ __forceinline__ v8f wmma_f16(v16h a, v16h b, v8f c) {
  return __builtin_amdgcn_wmma_f32_16x16x32_f16(
      /*neg_a=*/false, a, /*neg_b=*/false, b,
      /*c_mod=*/(short)0, c, /*reuse_a=*/false, /*reuse_b=*/false);
}

// per-lane slot -> matrix coordinate maps for v_wmma_f32_16x16x32_f16
__device__ __forceinline__ int a_row(int lane) { return lane & 15; }
__device__ __forceinline__ int a_col(int lane, int h, int kt) {
  // A (16x32 f16): lane<16: M=lane, K = h; lane>=16: K = h + 16
  return kt * 32 + h + ((lane >> 4) << 4);
}
__device__ __forceinline__ int b_row(int lane, int h, int kt) {
  // B (32x16 f16): lane<16: K = h; lane>=16: K = h + 16; N = lane&15
  return kt * 32 + h + ((lane >> 4) << 4);
}

// ---------------- weight prep: pack into f16 B-fragments ----------------
// fragment = 32 lanes x 16 halfs, contiguous (1KB); B[k][n], n = lane&15.
__global__ void prep_lin0(const float* __restrict__ W, h16* __restrict__ F) {
  const int t = blockIdx.x * blockDim.x + threadIdx.x;   // 32 frags * 32 lanes
  if (t >= 32 * 32) return;
  const int lane = t & 31, f = t >> 5;
  const int ot = f & 3, kt = (f >> 2) & 1, g = f >> 3;
  const int nn = lane & 15;
  h16* dst = F + (size_t)f * 512 + lane * 16;
#pragma unroll
  for (int h = 0; h < 16; ++h) {
    const int kk = b_row(lane, h, kt);
    const int o = ot * 16 + nn;
    float v = (kk < 33) ? W[(g * 64 + o) * 33 + kk] : 0.0f;
    dst[h] = (h16)v;
  }
}

__global__ void prep_lins(const float* __restrict__ W, h16* __restrict__ F) {
  const int t = blockIdx.x * blockDim.x + threadIdx.x;   // 96 frags * 32 lanes
  if (t >= 96 * 32) return;
  const int lane = t & 31, f = t >> 5;
  const int ot = f & 3; int r = f >> 2;
  const int kt = r & 1; r >>= 1;
  const int g = r & 3, l = r >> 2;
  const int nn = lane & 15;
  h16* dst = F + (size_t)f * 512 + lane * 16;
#pragma unroll
  for (int h = 0; h < 16; ++h) {
    const int kk = b_row(lane, h, kt);
    const int o = ot * 16 + nn;
    dst[h] = (h16)W[(((l * 4 + g) * 64) + o) * 64 + kk];
  }
}

__global__ void prep_gps(const float* __restrict__ W, h16* __restrict__ F) {
  const int t = blockIdx.x * blockDim.x + threadIdx.x;   // 640 frags * 32 lanes
  if (t >= 640 * 32) return;
  const int lane = t & 31, f = t >> 5;
  const int ot = f & 3; int r = f >> 2;
  const int kt = r & 1; r >>= 1;
  const int p = r % 20, l = r / 20;
  const int nn = lane & 15;
  h16* dst = F + (size_t)f * 512 + lane * 16;
#pragma unroll
  for (int h = 0; h < 16; ++h) {
    const int kk = b_row(lane, h, kt);          // input channel i
    const int o = ot * 16 + nn;                 // output channel
    dst[h] = (h16)W[((size_t)(l * 64 + o) * 64 + kk) * 20 + p];
  }
}

__global__ void prep_out(const float* __restrict__ W, h16* __restrict__ F) {
  const int t = blockIdx.x * blockDim.x + threadIdx.x;   // 512 frags * 32 lanes
  if (t >= 512 * 32) return;
  const int lane = t & 31, f = t >> 5;
  const int ot = f & 63; int r = f >> 6;
  const int kt = r & 1, g = r >> 1;
  const int nn = lane & 15;
  h16* dst = F + (size_t)f * 512 + lane * 16;
#pragma unroll
  for (int h = 0; h < 16; ++h) {
    const int kk = b_row(lane, h, kt);
    const int oc = ot * 16 + nn;                // 0..1023
    dst[h] = (h16)W[((size_t)g * 1024 + oc) * 64 + kk];
  }
}

// ---------------- phase-1 network kernel ----------------
// Activation LDS layout: [channel][n][blade(+1 pad)] -> conflict-light gathers.
__device__ __forceinline__ void lin_silu(
    float (*src)[16][9], float (*dst)[16][9],
    const h16* __restrict__ frags, const float* __restrict__ bias,
    const float* __restrict__ aA, const float* __restrict__ aB,
    int lane, int wave)
{
  float yv[8][8];
  const int row = a_row(lane);
  const int o = wave * 16 + (lane & 15);
#pragma unroll
  for (int bl = 0; bl < 8; ++bl) {
    v8f acc = {};
    const int g = CAY.grade[bl];
#pragma unroll
    for (int kt = 0; kt < 2; ++kt) {
      v16h a;
#pragma unroll
      for (int h = 0; h < 16; ++h)
        a[h] = (h16)src[a_col(lane, h, kt)][row][bl];
      const v16h b = *(const v16h*)(frags + (size_t)((g * 2 + kt) * 4 + wave) * 512 + lane * 16);
      acc = wmma_f16(a, b, acc);
    }
    const float bo = (bl == 0) ? bias[o] : 0.0f;
#pragma unroll
    for (int v = 0; v < 8; ++v) yv[bl][v] = acc[v] + bo;
  }
  float ga[4], gb[4];
#pragma unroll
  for (int g = 0; g < 4; ++g) { ga[g] = aA[o * 4 + g]; gb[g] = aB[o * 4 + g]; }
#pragma unroll
  for (int v = 0; v < 8; ++v) {
    const int nl = v + ((lane >> 4) << 3);
    float qg[4] = {0.f, 0.f, 0.f, 0.f};
#pragma unroll
    for (int bl = 0; bl < 8; ++bl) qg[CAY.grade[bl]] += yv[bl][v] * yv[bl][v];
    float gate[4];
#pragma unroll
    for (int g = 0; g < 4; ++g)
      gate[g] = 1.0f / (1.0f + __expf(-(ga[g] * qg[g] + gb[g])));
#pragma unroll
    for (int bl = 0; bl < 8; ++bl)
      dst[o][nl][bl] = yv[bl][v] * gate[CAY.grade[bl]];
  }
}

__device__ __forceinline__ void gp_layer(
    float (*src)[16][9], float (*dst)[16][9],
    const h16* __restrict__ gfrags, int lane, int wave)
{
  v8f acc[8];
#pragma unroll
  for (int j = 0; j < 8; ++j) acc[j] = v8f{};
  const int row = a_row(lane);
#pragma unroll
  for (int k = 0; k < 8; ++k) {
#pragma unroll
    for (int m = 0; m < 8; ++m) {
      const int   j = CAY.prod[k][m];
      const float s = CAY.sign[k][m];
      const int   p = CAY.pid[k][m];
#pragma unroll
      for (int kt = 0; kt < 2; ++kt) {
        v16h a;
#pragma unroll
        for (int h = 0; h < 16; ++h) {
          const int c = a_col(lane, h, kt);
          a[h] = (h16)(s * src[c][row][k] * src[c][row][m]);
        }
        const v16h b = *(const v16h*)(gfrags + (size_t)((p * 2 + kt) * 4 + wave) * 512 + lane * 16);
        acc[j] = wmma_f16(a, b, acc[j]);
      }
    }
  }
  const int o = wave * 16 + (lane & 15);
#pragma unroll
  for (int j = 0; j < 8; ++j) {
#pragma unroll
    for (int v = 0; v < 8; ++v) {
      const int nl = v + ((lane >> 4) << 3);
      dst[o][nl][j] = acc[j][v] * 0.125f;   // 1/sqrt(64)
    }
  }
}

__global__ __launch_bounds__(128) void net_kernel(
    const float* __restrict__ condition, const float* __restrict__ rel_pos_sigma,
    const float* __restrict__ lin0_b, const float* __restrict__ act0_a,
    const float* __restrict__ act0_b, const float* __restrict__ lins_b,
    const float* __restrict__ acts_a, const float* __restrict__ acts_b,
    const float* __restrict__ out_b, const float* __restrict__ shell_sigma,
    const h16* __restrict__ lin0F, const h16* __restrict__ linsF,
    const h16* __restrict__ gpsF, const h16* __restrict__ outF,
    float* __restrict__ kf_t)
{
  __shared__ float sx[HH][16][9];
  __shared__ float sy[HH][16][9];
  __shared__ float sq[16];
  const int tid = threadIdx.x, lane = tid & 31, wave = tid >> 5;
  const int n0 = blockIdx.x * 16;

  // zero activations + per-point q
  for (int idx = tid; idx < HH * 16 * 9; idx += 128) (&sx[0][0][0])[idx] = 0.0f;
  if (tid < 16) {
    const int n = n0 + tid;
    float q = 0.0f;
    if (n < NG) {
      const int iz = n % 7, iy = (n / 7) % 7, ix = n / 49;
      const float rx = (float)ix - 3.f, ry = (float)iy - 3.f, rz = (float)iz - 3.f;
      q = rx * rx + ry * ry + rz * rz;
    }
    sq[tid] = q;
  }
  const float sig = rel_pos_sigma[0];
  __syncthreads();

  // build x: channel 0 = exp(-q/2s^2) @blade0 + rel_pos @grade1; 1..32 = condition
  for (int idx = tid; idx < 16 * 33 * 8; idx += 128) {
    const int b = idx & 7;
    const int c = (idx >> 3) % 33;
    const int nl = idx / (33 * 8);
    const int n = n0 + nl;
    if (n >= NG) continue;
    float val;
    if (c == 0) {
      const int iz = n % 7, iy = (n / 7) % 7, ix = n / 49;
      const float rx = (float)ix - 3.f, ry = (float)iy - 3.f, rz = (float)iz - 3.f;
      const float q = rx * rx + ry * ry + rz * rz;
      const float sc = __expf(-q / (2.0f * sig * sig));
      val = (b == 0) ? sc : (b == 1) ? rx : (b == 2) ? ry : (b == 3) ? rz : 0.0f;
    } else {
      val = condition[(c - 1) * 8 + b];
    }
    sx[c][nl][b] = val;
  }
  __syncthreads();

  // 4 x (mv_linear + mv_silu + gp_layer); layer 0 weights are the K=33-padded set
#pragma unroll 1
  for (int l = 0; l < 4; ++l) {
    const h16*   lf = (l == 0) ? lin0F : (linsF + (size_t)(l - 1) * 32 * 512);
    const float* bb = (l == 0) ? lin0_b : (lins_b + (l - 1) * 64);
    const float* aa = (l == 0) ? act0_a : (acts_a + (l - 1) * 256);
    const float* ab = (l == 0) ? act0_b : (acts_b + (l - 1) * 256);
    lin_silu(sx, sy, lf, bb, aa, ab, lane, wave);
    __syncthreads();
    gp_layer(sy, sx, gpsF + (size_t)l * 160 * 512, lane, wave);
    __syncthreads();
  }

  // output projection: (16 x 64) x (64 x 1024) per blade, + bias/shell/FACTOR,
  // store transposed kf_t[(oc*8 + blade)*NPAD + n] for coalesced phase-2 reads
  const int row = a_row(lane);
#pragma unroll 1
  for (int ot8 = 0; ot8 < 16; ++ot8) {
    const int ot = wave * 16 + ot8;          // 0..63
    const int oc = ot * 16 + (lane & 15);    // 0..1023
#pragma unroll
    for (int bl = 0; bl < 8; ++bl) {
      v8f acc = {};
      const int g = CAY.grade[bl];
#pragma unroll
      for (int kt = 0; kt < 2; ++kt) {
        v16h a;
#pragma unroll
        for (int h = 0; h < 16; ++h)
          a[h] = (h16)sx[a_col(lane, h, kt)][row][bl];
        const v16h b = *(const v16h*)(outF + (size_t)((g * 2 + kt) * 64 + ot) * 512 + lane * 16);
        acc = wmma_f16(a, b, acc);
      }
      const float ssg = shell_sigma[oc * 8 + bl];
      const float inv_s2 = 1.0f / (ssg * ssg);
      const float bo = (bl == 0) ? out_b[oc] : 0.0f;
#pragma unroll
      for (int v = 0; v < 8; ++v) {
        const int nl = v + ((lane >> 4) << 3);
        const int n = n0 + nl;
        if (n < NG) {
          float val = (acc[v] + bo) * __expf(-sq[nl] * inv_s2) * FACTOR;
          kf_t[(size_t)(oc * 8 + bl) * NPAD + n] = val;
        }
      }
    }
  }
}

// ---------------- phase-2 expansion (bandwidth-bound) ----------------
// K[o,l,i,m,n] = kf[n,o,i,k] * cayley_w[o,i,pid(gk,gl,gm)] * sign, k = l XOR m
__global__ __launch_bounds__(256) void expand_kernel(
    const float* __restrict__ kf_t, const float* __restrict__ cayley_w,
    float* __restrict__ out)
{
  __shared__ float cw[20];
  const int oi = blockIdx.x;                 // o*32 + i
  const int o = oi >> 5, i = oi & 31;
  if (threadIdx.x < 20) cw[threadIdx.x] = cayley_w[oi * 20 + threadIdx.x];
  __syncthreads();
#pragma unroll 1
  for (int lm = 0; lm < 64; ++lm) {
    const int l = lm >> 3, m = lm & 7;
    const int k = CAY.m2i[CAY.mask[l] ^ CAY.mask[m]];
    const float w = cw[CAY.pid[k][m]] * CAY.sign[k][m];
    const float* srcp = kf_t + (size_t)(oi * 8 + k) * NPAD;
    float* dstp = out + (size_t)((o * 8 + l) * 256 + (i * 8 + m)) * 343;
    for (int n = threadIdx.x; n < 343; n += 256) dstp[n] = w * srcp[n];
  }
}

// ---------------- host launcher ----------------
extern "C" void kernel_launch(void* const* d_in, const int* in_sizes, int n_in,
                              void* d_out, int out_size, void* d_ws, size_t ws_size,
                              hipStream_t stream) {
  (void)in_sizes; (void)n_in; (void)out_size;
  const float* condition     = (const float*)d_in[0];
  const float* rel_pos_sigma = (const float*)d_in[1];
  const float* cayley_w      = (const float*)d_in[2];
  const float* lin0_W        = (const float*)d_in[3];
  const float* lin0_b        = (const float*)d_in[4];
  const float* act0_a        = (const float*)d_in[5];
  const float* act0_b        = (const float*)d_in[6];
  const float* gps_w         = (const float*)d_in[7];
  const float* lins_W        = (const float*)d_in[8];
  const float* lins_b        = (const float*)d_in[9];
  const float* acts_a        = (const float*)d_in[10];
  const float* acts_b        = (const float*)d_in[11];
  const float* out_W         = (const float*)d_in[12];
  const float* out_b         = (const float*)d_in[13];
  const float* shell_sigma   = (const float*)d_in[14];

  char* ws = (char*)d_ws;
  h16*   lin0F = (h16*)(ws + OFF_LIN0F);
  h16*   linsF = (h16*)(ws + OFF_LINSF);
  h16*   gpsF  = (h16*)(ws + OFF_GPSF);
  h16*   outF  = (h16*)(ws + OFF_OUTF);
  float* kf_t  = (float*)(ws + OFF_KFT);
  (void)ws_size;   // requires WS_NEED (~12.9 MB)

  prep_lin0<<<(32 * 32 + 127) / 128, 128, 0, stream>>>(lin0_W, lin0F);
  prep_lins<<<(96 * 32 + 127) / 128, 128, 0, stream>>>(lins_W, linsF);
  prep_gps <<<(640 * 32) / 128, 128, 0, stream>>>(gps_w, gpsF);
  prep_out <<<(512 * 32) / 128, 128, 0, stream>>>(out_W, outF);

  net_kernel<<<(NG + 15) / 16, 128, 0, stream>>>(
      condition, rel_pos_sigma, lin0_b, act0_a, act0_b,
      lins_b, acts_a, acts_b, out_b, shell_sigma,
      lin0F, linsF, gpsF, outF, kf_t);

  expand_kernel<<<1024, 256, 0, stream>>>(kf_t, cayley_w, (float*)d_out);
}
